// PropgationNet_1x_6588479832199
// MI455X (gfx1250) — matrix operand order repaired and checked by reading, not compiled, not run on hardware
//
#include <hip/hip_runtime.h>
#include <cstdint>

// ---------------------------------------------------------------------------
// Problem constants (from setup_inputs)
// ---------------------------------------------------------------------------
#define B_    8
#define C_IN  32          // guidance channels
#define C1    64          // conv1 output channels
#define H     80
#define W     160
#define HW    12800       // H*W
#define H2    160         // 2*H (guidance height)
#define W2    320         // 2*W (guidance width)
#define M2    576         // conv2 output channels (9*64)
#define K1    288         // conv1 GEMM K = 32*9
#define K2    576         // conv2 GEMM K = 64*9
#define BNT1  32          // spatial tile, kernel 1
#define BNT2  64          // spatial tile, kernel 2 (halves weight refetch)
#define EPS   1e-5f

typedef __attribute__((ext_vector_type(16))) __bf16 v16bf;
typedef __attribute__((ext_vector_type(8)))  float  v8f;

union FragU { uint4 q[2]; v16bf v; };

// ---------------------------------------------------------------------------
// CDNA5 async global->LDS copies (ASYNCcnt-tracked), with sync fallback.
// Builtin prototype (verified by toolchain diagnostic in compile probe):
//   void __builtin_amdgcn_global_load_async_to_lds_b128(
//       int4_vs addrspace(1)* src, int4_vs addrspace(3)* dst, imm offset, imm cpol)
// Generic->AS(1): same 64-bit value. Generic->AS(3): addr[31:0] is LDS offset.
// ---------------------------------------------------------------------------
typedef int v4i_vs __attribute__((vector_size(16)));   // matches builtin pointee

#if defined(__gfx1250__) && defined(__has_builtin)
#  if __has_builtin(__builtin_amdgcn_global_load_async_to_lds_b128) && \
      __has_builtin(__builtin_amdgcn_s_wait_asynccnt)
#    define HAVE_ASYNC_LDS 1
#  endif
#endif

#if defined(HAVE_ASYNC_LDS)
#  define COPY16(lds_dst, gsrc)                                                   \
      __builtin_amdgcn_global_load_async_to_lds_b128(                             \
          (__attribute__((address_space(1))) v4i_vs*)(uintptr_t)(gsrc),           \
          (__attribute__((address_space(3))) v4i_vs*)(unsigned)(uintptr_t)(lds_dst), \
          0, 0)
#  define AWAIT() __builtin_amdgcn_s_wait_asynccnt(0)
#else
#  define COPY16(lds_dst, gsrc) (*(uint4*)(lds_dst) = *(const uint4*)(gsrc))
#  define AWAIT() ((void)0)
#endif

__device__ __forceinline__ unsigned short f2bf(float f) {
    unsigned u = __float_as_uint(f);
    u += 0x7FFFu + ((u >> 16) & 1u);      // round-to-nearest-even
    return (unsigned short)(u >> 16);
}

// ---------------------------------------------------------------------------
// Kernel 0: repack conv weights fp32 -> bf16 GEMM layout [M][K=(tap,c)],
//           and fold BN into per-channel scale/bias.
// ---------------------------------------------------------------------------
__global__ __launch_bounds__(256)
void prep_weights(const float* __restrict__ w1, const float* __restrict__ w2,
                  const float* __restrict__ gamma, const float* __restrict__ beta,
                  const float* __restrict__ mean,  const float* __restrict__ var,
                  unsigned short* __restrict__ w1b, unsigned short* __restrict__ w2b,
                  float* __restrict__ bnscale, float* __restrict__ bnbias) {
    int idx = blockIdx.x * 256 + threadIdx.x;
    if (idx < C1 * K1) {
        int m = idx / K1, k = idx % K1;
        int tap = k / C_IN, c = k % C_IN;
        int ky = tap / 3, kx = tap % 3;
        w1b[idx] = f2bf(w1[((m * C_IN + c) * 3 + ky) * 3 + kx]);
        if (idx < C1) {                   // BN fold: y = x*scale + bias
            float sc = gamma[idx] * rsqrtf(var[idx] + EPS);
            bnscale[idx] = sc;
            bnbias[idx]  = beta[idx] - mean[idx] * sc;
        }
    } else {
        int i2 = idx - C1 * K1;
        if (i2 < M2 * K2) {
            int m = i2 / K2, k = i2 % K2;
            int tap = k / C1, c = k % C1;
            int ky = tap / 3, kx = tap % 3;
            w2b[i2] = f2bf(w2[((m * C1 + c) * 3 + ky) * 3 + kx]);
        }
    }
}

// ---------------------------------------------------------------------------
// Kernel 1: conv1 (stride2, pad1) + BN + ReLU  -> x bf16, NHWC layout
// Implicit GEMM: M=64, K=288 (9 taps * 32ch), N=32 spatial per WG.
// ---------------------------------------------------------------------------
__global__ __launch_bounds__(256)
void conv1_bn_relu(const float* __restrict__ g,
                   const float* __restrict__ bnscale, const float* __restrict__ bnbias,
                   const unsigned short* __restrict__ w1b,
                   unsigned short* __restrict__ x) {
    __shared__ __align__(16) unsigned char sm[C1 * K1 * 2 + BNT1 * 32 * 2]; // 36864 + 2048
    unsigned short* wt = (unsigned short*)sm;                  // [64][288] bf16
    unsigned short* at = (unsigned short*)(sm + C1 * K1 * 2);  // [32][32]  bf16

    const int t  = threadIdx.x;
    const int wg = blockIdx.x;
    const int b  = wg / (H * (W / BNT1));
    const int rm = wg % (H * (W / BNT1));
    const int h  = rm / (W / BNT1);
    const int w0 = (rm % (W / BNT1)) * BNT1;

    // cache all conv1 weights in LDS (2304 x 16B), async engine
    #pragma unroll
    for (int r = 0; r < 9; ++r) {
        const int idx = (t + 256 * r) * 16;
        COPY16((unsigned char*)wt + idx, (const unsigned char*)w1b + idx);
    }
    AWAIT();

    const int lane = t & 31;
    const int wv   = t >> 5;
    const int mt   = wv >> 1;        // 0..3  (M tile of 16)
    const int nt   = wv & 1;         // 0..1  (N tile of 16)
    const int lhi  = lane >> 4;
    const int llo  = lane & 15;

    const int n  = t & 31;           // spatial col this thread gathers
    const int cg = (t >> 5) << 2;    // channel group (4 ch)

    v8f acc = {};

    for (int kc = 0; kc < 9; ++kc) { // one 3x3 tap per K-chunk (BK = 32 = C_IN)
        __syncthreads();
        // branchless im2col gather: clamp address, unconditional load, select
        const int iy = kc / 3 - 1, jx = kc % 3 - 1;
        const int y  = 2 * h + iy;
        const int x2 = 2 * (w0 + n) + jx;
        const bool ok = (y >= 0) & (y < H2) & (x2 >= 0) & (x2 < W2);
        const int yc  = min(max(y, 0), H2 - 1);
        const int xc  = min(max(x2, 0), W2 - 1);
        const float* src = g + ((b * C_IN + cg) * H2 + yc) * W2 + xc;
        #pragma unroll
        for (int i = 0; i < 4; ++i) {
            float v = src[i * (H2 * W2)];
            v = ok ? v : 0.0f;
            at[n * 32 + cg + i] = f2bf(v);
        }
        __syncthreads();

        FragU a, bb;
        { // A (weights) 16x32: lanes<16 K{0..7,16..23}, lanes>=16 K{8..15,24..31}
            const unsigned char* row = (const unsigned char*)wt + (mt * 16 + llo) * (K1 * 2) + kc * 64;
            a.q[0] = *(const uint4*)(row + lhi * 16);
            a.q[1] = *(const uint4*)(row + 32 + lhi * 16);
        }
        { // B (acts) 32x16: lanes<16 K{0..7,16..23}, lanes>=16 K{8..15,24..31} for column llo
            const unsigned char* row = (const unsigned char*)at + (nt * 16 + llo) * 64;
            bb.q[0] = *(const uint4*)(row + lhi * 16);
            bb.q[1] = *(const uint4*)(row + 32 + lhi * 16);
        }
        acc = __builtin_amdgcn_wmma_f32_16x16x32_bf16(false, a.v, false, bb.v,
                                                      (short)0, acc, false, false);
    }

    // epilogue: folded BN + ReLU, pack 8 consecutive channels, one 16B store
    const int mBase   = mt * 16 + 8 * lhi;
    const int spatial = (b * H + h) * W + w0 + nt * 16 + llo;
    unsigned short pk[8];
    #pragma unroll
    for (int r = 0; r < 8; ++r) {
        const int m = mBase + r;
        float v = fmaf(acc[r], bnscale[m], bnbias[m]);
        v = v > 0.0f ? v : 0.0f;
        pk[r] = f2bf(v);
    }
    *(uint4*)(x + spatial * C1 + mBase) = *(const uint4*)pk;
}

// ---------------------------------------------------------------------------
// Kernel 2: conv2 (implicit GEMM, M=576, K=576, N-tile=64) + register-local
// softmax over 9 taps + convex-combination 8x upsample of 8*disp.
// 512 threads = 16 waves; wave wv: nt = wv&3 (N tile), mrow = wv>>2.
// Identity: acc[i][r] has m = 64*i + (mrow*16 + 8*lhi + r) => tap == i,
// sub == mrow*16+8*lhi+r => softmax in registers; lane owns 8 consecutive
// output pixels (dy = 2*mrow+lhi, dx = r).
// Double-buffered LDS tiles; async copies for tile k+1 overlap WMMA on tile k.
// Spatial tile = 64 consecutive flat indices (HW % 64 == 0, same batch).
// ---------------------------------------------------------------------------
#define TILE2_BYTES 40960              // 36864 weight + 4096 act per buffer

__global__ __launch_bounds__(512)
void conv2_softmax_up(const unsigned short* __restrict__ x,
                      const unsigned short* __restrict__ w2b,
                      const float* __restrict__ disp,
                      float* __restrict__ out) {
    __shared__ __align__(16) unsigned char sm[2 * TILE2_BYTES + BNT2 * 9 * 4];
    float* patches = (float*)(sm + 2 * TILE2_BYTES);      // [64][9]

    const int t   = threadIdx.x;
    const int wg  = blockIdx.x;
    const int b   = wg / (HW / BNT2);          // 200 tiles per batch image
    const int sl0 = (wg % (HW / BNT2)) * BNT2; // flat h*W+w of first column

    // unfold(8*disp, 3x3, pad 1) values for this spatial tile
    for (int idx = t; idx < BNT2 * 9; idx += 512) {
        const int nc = idx / 9, tap = idx % 9;
        const int sl = sl0 + nc;
        const int hn = sl / W, wn = sl % W;
        const int i = tap / 3, j = tap % 3;
        const int hh = hn + i - 1, ww = wn + j - 1;
        patches[idx] = (hh >= 0 && hh < H && ww >= 0 && ww < W)
                           ? 8.0f * disp[(b * H + hh) * W + ww] : 0.0f;
    }

    const int lane = t & 31;
    const int wv   = t >> 5;
    const int nt   = wv & 3;     // N tile (0..3)
    const int mrow = wv >> 2;    // M tiles: mrow + 4*i, i=0..8 (i == tap)
    const int lhi  = lane >> 4;
    const int llo  = lane & 15;

    // stage K-chunk kc into LDS buffer `buf` (async copy engine)
    auto stage = [&](int kc, int buf) {
        unsigned char* wtb = sm + buf * TILE2_BYTES;         // [576][32] bf16
        unsigned char* atb = wtb + 36864;                    // [64][32]  bf16
        // weight tile: 576x32 bf16 = 2304 x 16B, 4.5 per thread (512 thr)
        #pragma unroll
        for (int r = 0; r < 5; ++r) {
            const int idx = t + 512 * r;
            if (idx < 2304) {
                const int m = idx >> 2, q = idx & 3;
                COPY16(wtb + m * 64 + q * 16,
                       (const unsigned char*)w2b + m * (K2 * 2) + kc * 64 + q * 16);
            }
        }
        // activation tile: 64x32 bf16 = 256 x 16B; tap = kc>>1, c-half = kc&1
        if (t < 256) {
            const int n = t >> 2, q = t & 3;
            const int tap = kc >> 1;
            const int c0b = (kc & 1) * 64;
            const int iy = tap / 3 - 1, jx = tap % 3 - 1;
            const int sl = sl0 + n;
            const int hh = sl / W + iy, ww = sl % W + jx;
            unsigned char* dst = atb + n * 64 + q * 16;
            if (hh >= 0 && hh < H && ww >= 0 && ww < W) {
                const int spatial = (b * H + hh) * W + ww;
                COPY16(dst, (const unsigned char*)x + spatial * (C1 * 2) + c0b + q * 16);
            } else {
                *(uint4*)dst = make_uint4(0u, 0u, 0u, 0u);
            }
        }
    };

    v8f acc[9] = {};

    stage(0, 0);                            // prologue
    for (int kc = 0; kc < 18; ++kc) {       // K = 576, BK = 32
        const int cur = kc & 1;
        AWAIT();                            // my async copies into buf[cur] done
        __syncthreads();                    // tile visible; buf[cur^1] free
        if (kc + 1 < 18) stage(kc + 1, cur ^ 1);   // overlap with WMMA below

        const unsigned char* wtb = sm + cur * TILE2_BYTES;
        const unsigned char* atb = wtb + 36864;

        FragU bb;
        {
            const unsigned char* row = atb + (nt * 16 + llo) * 64;
            bb.q[0] = *(const uint4*)(row + lhi * 16);
            bb.q[1] = *(const uint4*)(row + 32 + lhi * 16);
        }
        #pragma unroll
        for (int i = 0; i < 9; ++i) {
            const int mtile = mrow + 4 * i;
            FragU a;
            const unsigned char* row = wtb + (mtile * 16 + llo) * 64;
            a.q[0] = *(const uint4*)(row + lhi * 16);
            a.q[1] = *(const uint4*)(row + 32 + lhi * 16);
            acc[i] = __builtin_amdgcn_wmma_f32_16x16x32_bf16(false, a.v, false, bb.v,
                                                             (short)0, acc[i], false, false);
        }
    }

    // --- register-local softmax over taps (acc[i] == tap i for this lane) ---
    const int ncol = nt * 16 + llo;      // spatial column 0..63
    float p[9];
    #pragma unroll
    for (int i = 0; i < 9; ++i) p[i] = patches[ncol * 9 + i];

    float outv[8];
    #pragma unroll
    for (int r = 0; r < 8; ++r) {
        float mx = acc[0][r];
        #pragma unroll
        for (int i = 1; i < 9; ++i) mx = fmaxf(mx, acc[i][r]);
        float s = 0.0f, up = 0.0f;
        #pragma unroll
        for (int i = 0; i < 9; ++i) {
            const float e = __expf(acc[i][r] - mx);
            s  += e;
            up += e * p[i];
        }
        outv[r] = up / s;
    }

    // lane owns 8 consecutive pixels: row 8h+dy, cols 8w .. 8w+7
    const int sl = sl0 + ncol;
    const int hn = sl / W, wn = sl % W;
    const int dy = 2 * mrow + lhi;
    float* dst = out + b * (8 * H * 8 * W) + (8 * hn + dy) * (8 * W) + 8 * wn;
    *(float4*)(dst)     = make_float4(outv[0], outv[1], outv[2], outv[3]);
    *(float4*)(dst + 4) = make_float4(outv[4], outv[5], outv[6], outv[7]);
}

// ---------------------------------------------------------------------------
extern "C" void kernel_launch(void* const* d_in, const int* in_sizes, int n_in,
                              void* d_out, int out_size, void* d_ws, size_t ws_size,
                              hipStream_t stream) {
    const float* guidance = (const float*)d_in[0];
    const float* disp     = (const float*)d_in[1];
    const float* conv1_w  = (const float*)d_in[2];
    const float* bn_gamma = (const float*)d_in[3];
    const float* bn_beta  = (const float*)d_in[4];
    const float* bn_mean  = (const float*)d_in[5];
    const float* bn_var   = (const float*)d_in[6];
    const float* conv2_w  = (const float*)d_in[7];

    unsigned char* ws = (unsigned char*)d_ws;
    unsigned short* w1b  = (unsigned short*)ws;                       // 36,864 B
    unsigned short* w2b  = (unsigned short*)(ws + 36864);             // 663,552 B
    unsigned short* xbuf = (unsigned short*)(ws + 36864 + 663552);    // 13,107,200 B
    float* bnscale = (float*)(ws + 36864 + 663552 + 13107200);        // 256 B
    float* bnbias  = (float*)(ws + 36864 + 663552 + 13107200 + 256);  // 256 B

    // 0: weight repack fp32->bf16 + BN fold (18432 + 331776 = 350208 elems)
    prep_weights<<<(18432 + 331776) / 256, 256, 0, stream>>>(
        conv1_w, conv2_w, bn_gamma, bn_beta, bn_mean, bn_var,
        w1b, w2b, bnscale, bnbias);

    // 1: conv1+BN+ReLU -> x (bf16 NHWC). 102400 spatial / 32 per WG = 3200 WGs.
    conv1_bn_relu<<<3200, 256, 0, stream>>>(guidance, bnscale, bnbias, w1b, xbuf);

    // 2: conv2 + register-local softmax + convex upsample -> out.
    //    102400 spatial / 64 per WG = 1600 WGs x 512 threads.
    conv2_softmax_up<<<1600, 512, 0, stream>>>(xbuf, w2b, disp, (float*)d_out);
}
